// AggrHGraphConvWindow_23089744183458
// MI455X (gfx1250) — matrix operand order repaired and checked by reading, not compiled, not run on hardware
//
#include <hip/hip_runtime.h>
#include <hip/hip_bf16.h>

typedef _Float16 half_t;
typedef __attribute__((ext_vector_type(16))) _Float16 v16h;
typedef __attribute__((ext_vector_type(8)))  _Float16 v8h;
typedef __attribute__((ext_vector_type(8)))  float    v8f;

#define N_NODE 500
#define N_POD  30000
#define N_SVC  3000
#define BTOT   33500
#define TT     16
#define FF     64
#define HH     128
#define G4     512

// Conv tiling over the concatenated row space [node | pod | svc]
#define TILES0 32     /* ceil(500/16)  */
#define TILES1 1875   /* 30000/16      */
#define TILES2 188    /* ceil(3000/16) */
#define CONV_TILES (TILES0 + TILES1 + TILES2)   /* 2095 */
#define LSTM_TILES ((BTOT + 15) / 16)           /* 2094 */

__device__ __forceinline__ float sigmoidf_(float x) {
    return 1.0f / (1.0f + __expf(-x));
}

// ---------------------------------------------------------------- utilities
__global__ void zero_f32_kernel(float* p, size_t n) {
    size_t i  = (size_t)blockIdx.x * blockDim.x + threadIdx.x;
    size_t st = (size_t)gridDim.x * blockDim.x;
    for (; i < n; i += st) p[i] = 0.0f;
}

__global__ void degree_kernel(const int* __restrict__ src, const int* __restrict__ dst,
                              int E, float* degOut, float* degIn) {
    int e = blockIdx.x * blockDim.x + threadIdx.x;
    if (e < E) {
        atomicAdd(&degOut[src[e]], 1.0f);
        atomicAdd(&degIn[dst[e]], 1.0f);
    }
}

// One block (256 threads) per edge: scatter x_src * rsqrt(deg_out) into agg[dst].
__global__ void scatter_kernel(const float* __restrict__ xsrc,
                               const int* __restrict__ src, const int* __restrict__ dst,
                               int E, const float* __restrict__ degOut,
                               float* __restrict__ aggDstBase) {
    int e = blockIdx.x;
    if (e >= E) return;
    int c = threadIdx.x;                       // 256 chunks of 4 floats over T*F=1024
    int s = src[e], d = dst[e];
    float scale = rsqrtf(fmaxf(degOut[s], 1.0f));
    const float4 v = ((const float4*)(xsrc + (long long)s * (TT * FF)))[c];
    float* a = aggDstBase + (long long)d * (TT * FF) + c * 4;
    atomicAdd(a + 0, v.x * scale);
    atomicAdd(a + 1, v.y * scale);
    atomicAdd(a + 2, v.z * scale);
    atomicAdd(a + 3, v.w * scale);
}

// ---------------------------------------------------------------- weight packing
// Conv weights (T,F,H) f32 -> f16 WMMA B-frags: [t][nt(8)][kc(2)][lane(32)][j(16)]
// B element (k,n): lane gives n = nt*16 + lane%16, half j gives k = kc*32 + (j&7) + (lane<16?0:8) + (j<8?0:16).
__global__ void pack_convW_kernel(const float* __restrict__ W, half_t* __restrict__ out) {
    int idx = blockIdx.x * blockDim.x + threadIdx.x;
    if (idx >= TT * 8 * 2 * 32 * 16) return;
    int j    =  idx        & 15;
    int lane = (idx >> 4)  & 31;
    int kc   = (idx >> 9)  & 1;
    int nt   = (idx >> 10) & 7;
    int t    =  idx >> 13;
    int n = nt * 16 + (lane & 15);
    int k = kc * 32 + (j & 7) + ((lane < 16) ? 0 : 8) + ((j < 8) ? 0 : 16);
    out[idx] = (half_t)W[((long long)t * FF + k) * HH + n];
}

// LSTM weights (512,128) f32; GEMM B-matrix is W^T: B(k,n)=W[n][k].
// Layout: [nt(32)][kc(4)][lane(32)][j(16)]
__global__ void pack_lstmW_kernel(const float* __restrict__ W, half_t* __restrict__ out) {
    int idx = blockIdx.x * blockDim.x + threadIdx.x;
    if (idx >= 32 * 4 * 32 * 16) return;
    int j    =  idx        & 15;
    int lane = (idx >> 4)  & 31;
    int kc   = (idx >> 9)  & 3;
    int nt   =  idx >> 11;
    int n = nt * 16 + (lane & 15);
    int k = kc * 32 + (j & 7) + ((lane < 16) ? 0 : 8) + ((j < 8) ? 0 : 16);
    out[idx] = (half_t)W[n * HH + k];
}

__global__ void bias_combine_kernel(const float* __restrict__ a, const float* __restrict__ b,
                                    float* __restrict__ out, int n) {
    int i = blockIdx.x * blockDim.x + threadIdx.x;
    if (i < n) out[i] = a[i] + b[i];
}

// ---------------------------------------------------------------- conv GEMM
// Grid: (CONV_TILES, T). Block: 256 (8 waves); wave w does N-tile w (H=128 -> 8 tiles).
__global__ void conv_gemm_kernel(const float* __restrict__ agg,
                                 const float* __restrict__ degIn_node,
                                 const float* __restrict__ degIn_pod,
                                 const float* __restrict__ degIn_svc,
                                 const half_t* __restrict__ wf_in,
                                 const half_t* __restrict__ wf_ni,
                                 const half_t* __restrict__ wf_svc,
                                 const float* __restrict__ b_in,
                                 const float* __restrict__ b_ni,
                                 const float* __restrict__ b_svc,
                                 half_t* __restrict__ X) {
    int tile = blockIdx.x;
    int t    = blockIdx.y;
    int base, secBase, secEnd;
    const half_t* wf; const float* bias; const float* degIn;
    if (tile < TILES0) {
        secBase = 0;      secEnd = N_NODE;          base = tile * 16;
        wf = wf_in;  bias = b_in;  degIn = degIn_node;
    } else if (tile < TILES0 + TILES1) {
        secBase = N_NODE; secEnd = N_NODE + N_POD;  base = secBase + (tile - TILES0) * 16;
        wf = wf_ni;  bias = b_ni;  degIn = degIn_pod;
    } else {
        secBase = N_NODE + N_POD; secEnd = BTOT;    base = secBase + (tile - TILES0 - TILES1) * 16;
        wf = wf_svc; bias = b_svc; degIn = degIn_svc;
    }
    int lane = threadIdx.x & 31;
    int wave = threadIdx.x >> 5;
    int mrow = lane & 15;
    int row  = base + mrow;
    bool valid = row < secEnd;
    float scale = valid ? rsqrtf(fmaxf(degIn[row - secBase], 1.0f)) : 0.0f;

    // A frags (16x32 f16 layout: lanes 0-15 hold K{0-7,16-23}, lanes 16-31 hold K{8-15,24-31})
    const float* arow = agg + (long long)row * (TT * FF) + t * FF;
    int kb = (lane < 16) ? 0 : 8;
    v16h a0, a1;
#pragma unroll
    for (int j = 0; j < 8; j++) {
        float x0 = valid ? arow[kb + j] : 0.0f;
        float x1 = valid ? arow[16 + kb + j] : 0.0f;
        float y0 = valid ? arow[32 + kb + j] : 0.0f;
        float y1 = valid ? arow[48 + kb + j] : 0.0f;
        a0[j] = (half_t)(x0 * scale);  a0[8 + j] = (half_t)(x1 * scale);
        a1[j] = (half_t)(y0 * scale);  a1[8 + j] = (half_t)(y1 * scale);
    }

    const v16h* wv = (const v16h*)wf;   // frag i at half-offset i*16
    v16h bw0 = wv[((t * 8 + wave) * 2 + 0) * 32 + lane];
    v16h bw1 = wv[((t * 8 + wave) * 2 + 1) * 32 + lane];

    v8f acc = {};
    acc = __builtin_amdgcn_wmma_f32_16x16x32_f16(false, a0, false, bw0, (short)0, acc, false, false);
    acc = __builtin_amdgcn_wmma_f32_16x16x32_f16(false, a1, false, bw1, (short)0, acc, false, false);

    float bv = bias[t * HH + wave * 16 + mrow];
    int rbase = (lane < 16) ? 0 : 8;
    long long colOff = (long long)t * HH + wave * 16 + mrow;
    if (base + 16 <= secEnd) {            // full tile: straight-line stores
#pragma unroll
        for (int r = 0; r < 8; r++) {
            float v = acc[r] + bv;
            v = (v > 0.0f) ? v : 0.01f * v;
            X[(long long)(base + rbase + r) * (TT * HH) + colOff] = (half_t)v;
        }
    } else {                              // boundary tile
#pragma unroll
        for (int r = 0; r < 8; r++) {
            int orow = base + rbase + r;
            if (orow < secEnd) {
                float v = acc[r] + bv;
                v = (v > 0.0f) ? v : 0.01f * v;
                X[(long long)orow * (TT * HH) + colOff] = (half_t)v;
            }
        }
    }
}

// ---------------------------------------------------------------- fused 2-layer LSTM
// One block = 16 batch rows for ALL T steps and BOTH layers.
// Per step per wave per layer: 4 nt x 4 kc x 2 matrices = 32 WMMAs (64 total).
__device__ __forceinline__ void load_frags_lds(const half_t (*buf)[HH], int mrow, int kb,
                                               v16h (&a)[4]) {
    const v8h* r = (const v8h*)&buf[mrow][0];
#pragma unroll
    for (int kc = 0; kc < 4; kc++) {
        v8h lo = r[(kc * 32 + kb) >> 3];
        v8h hi = r[(kc * 32 + 16 + kb) >> 3];
        a[kc] = __builtin_shufflevector(lo, hi, 0,1,2,3,4,5,6,7,8,9,10,11,12,13,14,15);
    }
}

__device__ __forceinline__ void gate_gemm(const v16h (&xa)[4], const v16h (&ha)[4],
                                          const v16h* __restrict__ wiv,
                                          const v16h* __restrict__ whv,
                                          const float* __restrict__ bias,
                                          float (*gates)[G4], int lane, int wave) {
    int mrow  = lane & 15;
    int rbase = (lane < 16) ? 0 : 8;
#pragma unroll
    for (int q = 0; q < 4; q++) {
        int nt = wave * 4 + q;
        v8f acc = {};
#pragma unroll
        for (int kc = 0; kc < 4; kc++) {
            acc = __builtin_amdgcn_wmma_f32_16x16x32_f16(
                    false, xa[kc], false, wiv[(nt * 4 + kc) * 32 + lane],
                    (short)0, acc, false, false);
            acc = __builtin_amdgcn_wmma_f32_16x16x32_f16(
                    false, ha[kc], false, whv[(nt * 4 + kc) * 32 + lane],
                    (short)0, acc, false, false);
        }
        float bv = bias[nt * 16 + mrow];
#pragma unroll
        for (int r = 0; r < 8; r++)
            gates[rbase + r][nt * 16 + mrow] = acc[r] + bv;
    }
}

// Cell nonlinearity for this thread's 8 channels; h returned in hv, written to hb.
__device__ __forceinline__ void cell_phase(const float (*gates)[G4], float* c,
                                           half_t (*hb)[HH], int row_cell, int col0,
                                           float* hv) {
    v8h hpack;
#pragma unroll
    for (int u = 0; u < 8; u++) {
        int j = col0 + u;
        float ig = gates[row_cell][j];
        float fg = gates[row_cell][HH + j];
        float gg = gates[row_cell][2 * HH + j];
        float og = gates[row_cell][3 * HH + j];
        float cs = sigmoidf_(fg) * c[u] + sigmoidf_(ig) * tanhf(gg);
        c[u] = cs;
        float h = sigmoidf_(og) * tanhf(cs);
        hv[u] = h;
        hpack[u] = (half_t)h;
    }
    *(v8h*)&hb[row_cell][col0] = hpack;   // single ds_store_b128
}

__global__ void lstm2_fused_kernel(const half_t* __restrict__ Xin,
                                   const half_t* __restrict__ wih0F,
                                   const half_t* __restrict__ whh0F,
                                   const float*  __restrict__ bias0,
                                   const half_t* __restrict__ wih1F,
                                   const half_t* __restrict__ whh1F,
                                   const float*  __restrict__ bias1,
                                   float* __restrict__ out) {
    __shared__ float gates[16][G4];
    __shared__ __align__(16) half_t hbuf0[16][HH];
    __shared__ __align__(16) half_t hbuf1[16][HH];

    int tid  = threadIdx.x;
    int lane = tid & 31;
    int wave = tid >> 5;
    int mrow = lane & 15;
    int rowbase = blockIdx.x * 16;
    long long grow = rowbase + mrow;
    bool valid = grow < BTOT;

    for (int i = tid; i < 16 * HH; i += 256) {
        ((half_t*)hbuf0)[i] = (half_t)0.0f;
        ((half_t*)hbuf1)[i] = (half_t)0.0f;
    }
    float c0[8], c1[8];
#pragma unroll
    for (int u = 0; u < 8; u++) { c0[u] = 0.0f; c1[u] = 0.0f; }
    int row_cell = tid >> 4;            // 0..15
    int col0     = (tid & 15) * 8;      // 8 channels per thread
    long long orow = rowbase + row_cell;
    bool ovalid = orow < BTOT;

    __builtin_prefetch(wih0F, 0, 1);
    __builtin_prefetch(whh0F, 0, 1);
    __builtin_prefetch(wih1F, 0, 1);
    __builtin_prefetch(whh1F, 0, 1);

    __syncthreads();

    const v16h* wi0 = (const v16h*)wih0F;
    const v16h* wh0 = (const v16h*)whh0F;
    const v16h* wi1 = (const v16h*)wih1F;
    const v16h* wh1 = (const v16h*)whh1F;
    int kb = (lane < 16) ? 0 : 8;
    const v8h* xr = (const v8h*)(Xin + grow * (TT * HH));

    for (int t = 0; t < TT; t++) {
        // ---- layer 0: gates = x_t*Wih0^T + h0*Whh0^T + bias0
        v16h xa[4], ha[4];
#pragma unroll
        for (int kc = 0; kc < 4; kc++) {
            v8h lo = {}, hi = {};
            if (valid) {
                lo = xr[(t * HH + kc * 32 + kb) >> 3];
                hi = xr[(t * HH + kc * 32 + 16 + kb) >> 3];
            }
            xa[kc] = __builtin_shufflevector(lo, hi, 0,1,2,3,4,5,6,7,8,9,10,11,12,13,14,15);
        }
        load_frags_lds(hbuf0, mrow, kb, ha);
        gate_gemm(xa, ha, wi0, wh0, bias0, gates, lane, wave);
        __syncthreads();

        float h0v[8];
        cell_phase(gates, c0, hbuf0, row_cell, col0, h0v);
        __syncthreads();

        // ---- layer 1: gates = h0_t*Wih1^T + h1*Whh1^T + bias1
        load_frags_lds(hbuf0, mrow, kb, xa);   // layer-1 input = layer-0 output
        load_frags_lds(hbuf1, mrow, kb, ha);
        gate_gemm(xa, ha, wi1, wh1, bias1, gates, lane, wave);
        __syncthreads();

        float h1v[8];
        cell_phase(gates, c1, hbuf1, row_cell, col0, h1v);
        if (ovalid) {                           // vector f32 stores (2 x b128)
            float4* op = (float4*)(out + orow * (TT * HH) + t * HH + col0);
            op[0] = make_float4(h1v[0], h1v[1], h1v[2], h1v[3]);
            op[1] = make_float4(h1v[4], h1v[5], h1v[6], h1v[7]);
        }
        __syncthreads();
    }
}

// ---------------------------------------------------------------- launch
extern "C" void kernel_launch(void* const* d_in, const int* in_sizes, int n_in,
                              void* d_out, int out_size, void* d_ws, size_t ws_size,
                              hipStream_t stream) {
    const float* node_feat = (const float*)d_in[0];
    const float* pod_feat  = (const float*)d_in[1];
    const float* svc_feat  = (const float*)d_in[2];
    const float* W_svc = (const float*)d_in[3];
    const float* b_svc = (const float*)d_in[4];
    const float* W_in  = (const float*)d_in[5];
    const float* b_in  = (const float*)d_in[6];
    const float* W_ni  = (const float*)d_in[7];
    const float* b_ni  = (const float*)d_in[8];
    const float* Wih0 = (const float*)d_in[9];
    const float* Whh0 = (const float*)d_in[10];
    const float* bih0 = (const float*)d_in[11];
    const float* bhh0 = (const float*)d_in[12];
    const float* Wih1 = (const float*)d_in[13];
    const float* Whh1 = (const float*)d_in[14];
    const float* bih1 = (const float*)d_in[15];
    const float* bhh1 = (const float*)d_in[16];
    const int* svc_src = (const int*)d_in[17];
    const int* svc_dst = (const int*)d_in[18];
    const int* in_src  = (const int*)d_in[19];   // pod -> node
    const int* in_dst  = (const int*)d_in[20];
    const int* ni_src  = (const int*)d_in[21];   // node -> pod
    const int* ni_dst  = (const int*)d_in[22];
    float* out = (float*)d_out;

    // ---- workspace layout (256B aligned slabs) ----
    char* w = (char*)d_ws;
    size_t off = 0;
    auto take = [&](size_t bytes) -> char* {
        char* p = w + off;
        off = (off + bytes + 255) & ~(size_t)255;
        return p;
    };
    float* degIn_node  = (float*)take(N_NODE * 4);
    float* degOut_pod  = (float*)take(N_POD * 4);
    float* degIn_pod   = (float*)take(N_POD * 4);
    float* degOut_node = (float*)take(N_NODE * 4);
    float* degIn_svc   = (float*)take(N_SVC * 4);
    float* degOut_svc  = (float*)take(N_SVC * 4);
    float* agg         = (float*)take((size_t)BTOT * TT * FF * 4);
    size_t zero_floats = off / 4;   // zero all degree buffers + agg every call
    half_t* wf_in  = (half_t*)take(TT * 8 * 2 * 32 * 16 * 2);
    half_t* wf_ni  = (half_t*)take(TT * 8 * 2 * 32 * 16 * 2);
    half_t* wf_svc = (half_t*)take(TT * 8 * 2 * 32 * 16 * 2);
    half_t* wf_ih0 = (half_t*)take(32 * 4 * 32 * 16 * 2);
    half_t* wf_hh0 = (half_t*)take(32 * 4 * 32 * 16 * 2);
    half_t* wf_ih1 = (half_t*)take(32 * 4 * 32 * 16 * 2);
    half_t* wf_hh1 = (half_t*)take(32 * 4 * 32 * 16 * 2);
    float* bias0 = (float*)take(G4 * 4);
    float* bias1 = (float*)take(G4 * 4);
    half_t* X  = (half_t*)take((size_t)BTOT * TT * HH * 2);
    (void)ws_size; (void)n_in; (void)in_sizes; (void)out_size;

    // 1) zero degree + agg scratch
    zero_f32_kernel<<<8192, 256, 0, stream>>>((float*)d_ws, zero_floats);

    // 2) degrees
    degree_kernel<<<(30000 + 255) / 256, 256, 0, stream>>>(in_src, in_dst, 30000, degOut_pod, degIn_node);
    degree_kernel<<<(30000 + 255) / 256, 256, 0, stream>>>(ni_src, ni_dst, 30000, degOut_node, degIn_pod);
    degree_kernel<<<(12000 + 255) / 256, 256, 0, stream>>>(svc_src, svc_dst, 12000, degOut_svc, degIn_svc);

    // 3) scatter (agg row space: [node 0..499 | pod 500..30499 | svc 30500..33499])
    scatter_kernel<<<30000, 256, 0, stream>>>(pod_feat,  in_src,  in_dst,  30000, degOut_pod,
                                              agg);
    scatter_kernel<<<30000, 256, 0, stream>>>(node_feat, ni_src,  ni_dst,  30000, degOut_node,
                                              agg + (size_t)N_NODE * TT * FF);
    scatter_kernel<<<12000, 256, 0, stream>>>(svc_feat,  svc_src, svc_dst, 12000, degOut_svc,
                                              agg + (size_t)(N_NODE + N_POD) * TT * FF);

    // 4) pack weights into WMMA B-fragment layouts
    const int convWN = TT * 8 * 2 * 32 * 16;
    pack_convW_kernel<<<convWN / 256, 256, 0, stream>>>(W_in,  wf_in);
    pack_convW_kernel<<<convWN / 256, 256, 0, stream>>>(W_ni,  wf_ni);
    pack_convW_kernel<<<convWN / 256, 256, 0, stream>>>(W_svc, wf_svc);
    const int lstmWN = 32 * 4 * 32 * 16;
    pack_lstmW_kernel<<<lstmWN / 256, 256, 0, stream>>>(Wih0, wf_ih0);
    pack_lstmW_kernel<<<lstmWN / 256, 256, 0, stream>>>(Whh0, wf_hh0);
    pack_lstmW_kernel<<<lstmWN / 256, 256, 0, stream>>>(Wih1, wf_ih1);
    pack_lstmW_kernel<<<lstmWN / 256, 256, 0, stream>>>(Whh1, wf_hh1);
    bias_combine_kernel<<<2, 256, 0, stream>>>(bih0, bhh0, bias0, G4);
    bias_combine_kernel<<<2, 256, 0, stream>>>(bih1, bhh1, bias1, G4);

    // 5) graph-conv linear + bias + LeakyReLU -> f16 X
    conv_gemm_kernel<<<dim3(CONV_TILES, TT), 256, 0, stream>>>(
        agg, degIn_node, degIn_pod, degIn_svc,
        wf_in, wf_ni, wf_svc, b_in, b_ni, b_svc, X);

    // 6) fused 2-layer LSTM -> f32 d_out (no inter-layer HBM round trip)
    lstm2_fused_kernel<<<LSTM_TILES, 256, 0, stream>>>(
        X, wf_ih0, wf_hh0, bias0, wf_ih1, wf_hh1, bias1, out);
}
